// S4Layer_39659728011555
// MI455X (gfx1250) — matrix operand order repaired and checked by reading, not compiled, not run on hardware
//
#include <hip/hip_runtime.h>
#include <hip/hip_bf16.h>

typedef __attribute__((ext_vector_type(16))) __bf16 v16bf;
typedef __attribute__((ext_vector_type(8)))  float  v8f;
typedef __attribute__((ext_vector_type(4)))  int    v4i;

#define BATCH  4
#define SEQ    2048
#define DMODEL 512
#define DSTATE 64
#define LN_EPS 1e-5f

#define WMMA_BF16(A, B, C) \
  __builtin_amdgcn_wmma_f32_16x16x32_bf16(false, (A), false, (B), (short)0, (C), false, false)

#if defined(__has_builtin)
#  if __has_builtin(__builtin_amdgcn_global_load_async_to_lds_b128)
#    define HAVE_ASYNC_LDS 1
#  endif
#endif
#ifndef HAVE_ASYNC_LDS
#  define HAVE_ASYNC_LDS 0
#endif

// ---------------- Kernel 0: W_out f32 -> bf16 ----------------
__global__ void wcvt_kernel(const float* __restrict__ W, __bf16* __restrict__ Wbf, int n) {
  int i = blockIdx.x * blockDim.x + threadIdx.x;
  if (i < n) Wbf[i] = (__bf16)W[i];
}

// ---------------- Kernel 1: SSM kernel K[d,l] (bf16) ----------------
// K[d,l] = sum_s exp(l*Ar) * (P*cos(l*Ai) - Q*sin(l*Ai)),
// Ar = -exp(log_A_real), P = CrBr+CiBi, Q = CrBi-CiBr.
__global__ void ssmk_kernel(const float* __restrict__ lar, const float* __restrict__ aim,
                            const float* __restrict__ Bp,  const float* __restrict__ Cp,
                            __bf16* __restrict__ Ksbf) {
  int d = blockIdx.x;
  __shared__ float Ar[DSTATE], Ai[DSTATE], P[DSTATE], Q[DSTATE];
  int tid = threadIdx.x;
  if (tid < DSTATE) {
    int s = tid;
    float br = Bp[(d*DSTATE+s)*2+0], bi = Bp[(d*DSTATE+s)*2+1];
    float cr = Cp[(d*DSTATE+s)*2+0], ci = Cp[(d*DSTATE+s)*2+1];
    Ar[s] = -__expf(lar[d*DSTATE+s]);
    Ai[s] = aim[d*DSTATE+s];
    P[s]  = cr*br + ci*bi;
    Q[s]  = cr*bi - ci*br;
  }
  __syncthreads();
  for (int l = tid; l < SEQ; l += blockDim.x) {
    float lf = (float)l, acc = 0.f;
    #pragma unroll 4
    for (int s = 0; s < DSTATE; ++s) {
      float e = __expf(lf * Ar[s]);
      float sn, cs;
      __sincosf(lf * Ai[s], &sn, &cs);
      acc += e * (P[s]*cs - Q[s]*sn);
    }
    Ksbf[d*SEQ + l] = (__bf16)acc;
  }
}

// ---------------- Kernel 2: block-Toeplitz depthwise conv via WMMA ----------------
// y1[b,t,d] = sum_{j>=0} K[d,j]*x[b,t+j,d] + D[d]*x[b,t,d]
// y_blk[I] = sum_Δ T_Δ @ x_blk[I+Δ],  T_Δ[i,j] = K[16Δ + j - i].
// Two independent accumulators (Δ, Δ+2) break the bf16 WMMA->WMMA RAW chain;
// zero-padded LDS tails make the overshoot iteration contribute exactly zero,
// so no divergent guard is needed and EXEC stays all-ones around every WMMA.
#define XS_LEN (SEQ + 320)             // back zero-pad (t beyond SEQ, incl. overshoot)
#define KS_PAD 32
#define KS_LEN (SEQ + 96)              // front pad 32 (j-i<0) + back pad 64

__global__ void conv_kernel(const float* __restrict__ xg, const __bf16* __restrict__ Ksbf,
                            const float* __restrict__ Dp, __bf16* __restrict__ y1bf) {
  int d = blockIdx.x, b = blockIdx.y;
  __shared__ __bf16 xs[XS_LEN];
  __shared__ __bf16 ks[KS_LEN];
  int tid = threadIdx.x;

#if HAVE_ASYNC_LDS
  // K row is already bf16 and contiguous: raw async DMA into LDS (256 lanes x 16B = 4 KiB).
  if (tid < KS_PAD) ks[tid] = (__bf16)0.f;
  if (tid < KS_LEN - KS_PAD - SEQ) ks[KS_PAD + SEQ + tid] = (__bf16)0.f;
  {
    const __bf16* src = Ksbf + (size_t)d * SEQ + (size_t)tid * 8;  // 8 bf16 = 16B/lane
    __bf16* dst = &ks[KS_PAD] + tid * 8;
    __builtin_amdgcn_global_load_async_to_lds_b128(
        (__attribute__((address_space(1))) v4i*)(uintptr_t)src,
        (__attribute__((address_space(3))) v4i*)dst, 0, 0);
  }
#else
  for (int i = tid; i < KS_LEN; i += blockDim.x) {
    int k = i - KS_PAD;
    ks[i] = (k >= 0 && k < SEQ) ? Ksbf[d*SEQ + k] : (__bf16)0.f;
  }
#endif
  // x is strided f32 -> needs conversion, stage through VALU.
  for (int i = tid; i < XS_LEN; i += blockDim.x)
    xs[i] = (i < SEQ) ? (__bf16)xg[((size_t)b*SEQ + i)*DMODEL + d] : (__bf16)0.f;
#if HAVE_ASYNC_LDS
  asm volatile("s_wait_asynccnt 0" ::: "memory");
#endif
  __syncthreads();

  int lane = tid & 31, wid = tid >> 5;
  int T0 = wid * 256;                  // each wave owns 256 outputs (16 blocks = N dim)
  int m  = lane & 15;                  // A row (== B column n under wave32 striping)
  int hi = lane >> 4;                  // half-wave selector (contiguous K split: k = 16*hi + e)
  v8f c0 = {}, c1 = {};
  for (int dlt = 0; T0 + 16*dlt < SEQ; dlt += 4) {
    {
      v16bf a, bm;
      int kb = 16*dlt + KS_PAD - m;
      #pragma unroll
      for (int e = 0; e < 8; ++e) { a[e] = ks[kb + hi*16 + e]; a[8+e] = ks[kb + hi*16 + 8 + e]; }
      int tb = T0 + 16*m + 16*dlt + hi*16;
      #pragma unroll
      for (int e = 0; e < 16; ++e) bm[e] = xs[tb + e];
      c0 = WMMA_BF16(a, bm, c0);
    }
    {
      v16bf a, bm;
      int kb = 16*(dlt + 2) + KS_PAD - m;
      #pragma unroll
      for (int e = 0; e < 8; ++e) { a[e] = ks[kb + hi*16 + e]; a[8+e] = ks[kb + hi*16 + 8 + e]; }
      int tb = T0 + 16*m + 16*(dlt + 2) + hi*16;
      #pragma unroll
      for (int e = 0; e < 16; ++e) bm[e] = xs[tb + e];
      c1 = WMMA_BF16(a, bm, c1);
    }
  }
  v8f c = c0 + c1;
  // C/D layout: lane -> N = lane&15, VGPR r -> M = r + 8*(lane>>4); t = T0 + 16N + M
  int n = lane & 15;
  float dd = Dp[d];
  #pragma unroll
  for (int r = 0; r < 8; ++r) {
    int t = T0 + 16*n + r + 8*hi;
    size_t gi = ((size_t)b*SEQ + t)*DMODEL + d;
    y1bf[gi] = (__bf16)(c[r] + dd * xg[gi]);
  }
}

// ---------------- Kernel 3: out-projection GEMM + bias + residual ----------------
// z[tok,o] = sum_k y1[tok,k]*W[o,k] + b_out[o] + x[tok,o]
// Each wave computes a 32x64 tile as 2x4 WMMA sub-tiles: A reused 4x, B reused 2x,
// 8 independent accumulators (no WMMA RAW hazard stalls).
__global__ void gemm_kernel(const __bf16* __restrict__ y1bf, const __bf16* __restrict__ Wbf,
                            const float* __restrict__ bo, const float* __restrict__ xg,
                            float* __restrict__ z) {
  int tid  = threadIdx.x;
  int lane = tid & 31;
  int wg   = blockIdx.x * (blockDim.x >> 5) + (tid >> 5);
  int mt = wg >> 3;                    // 0..255 : 32-token tiles
  int nt = wg & 7;                     // 0..7   : 64-output tiles
  int tok0 = mt * 32, o0 = nt * 64;
  int lo = lane & 15, hi = lane >> 4;
  const __bf16* a0r = y1bf + (size_t)(tok0 + lo) * DMODEL + hi*8;
  const __bf16* a1r = a0r + (size_t)16 * DMODEL;
  const __bf16* b0r = Wbf  + (size_t)(o0  + lo) * DMODEL + hi*8;
  v8f c[2][4] = {};
  for (int kb = 0; kb < DMODEL; kb += 32) {
    __builtin_prefetch(a0r + kb + 128, 0, 1);   // global_prefetch_b8, 256B ahead
    __builtin_prefetch(b0r + kb + 128, 0, 1);
    v16bf fa[2], fb[4];
    #pragma unroll
    for (int e = 0; e < 8; ++e) {
      fa[0][e] = a0r[kb + e]; fa[0][8+e] = a0r[kb + 16 + e];
      fa[1][e] = a1r[kb + e]; fa[1][8+e] = a1r[kb + 16 + e];
    }
    #pragma unroll
    for (int j = 0; j < 4; ++j) {
      const __bf16* bj = b0r + (size_t)j * 16 * DMODEL + kb;
      #pragma unroll
      for (int e = 0; e < 8; ++e) { fb[j][e] = bj[e]; fb[j][8+e] = bj[16 + e]; }
    }
    #pragma unroll
    for (int mi = 0; mi < 2; ++mi)
      #pragma unroll
      for (int j = 0; j < 4; ++j)
        c[mi][j] = WMMA_BF16(fa[mi], fb[j], c[mi][j]);
  }
  #pragma unroll
  for (int j = 0; j < 4; ++j) {
    int o = o0 + j*16 + lo;
    float bb = bo[o];
    #pragma unroll
    for (int mi = 0; mi < 2; ++mi) {
      #pragma unroll
      for (int r = 0; r < 8; ++r) {
        int tok = tok0 + mi*16 + r + 8*hi;
        size_t gi = (size_t)tok * DMODEL + o;
        z[gi] = c[mi][j][r] + bb + xg[gi];
      }
    }
  }
}

// ---------------- Kernel 4: LayerNorm over d=512 ----------------
__global__ void ln_kernel(const float* __restrict__ z, const float* __restrict__ gamma,
                          const float* __restrict__ beta, float* __restrict__ out) {
  int tok = blockIdx.x, tid = threadIdx.x;
  __shared__ float s1[256], s2[256];
  const float* zr = z + (size_t)tok * DMODEL;
  float v0 = zr[tid], v1 = zr[tid + 256];
  s1[tid] = v0 + v1;
  s2[tid] = v0*v0 + v1*v1;
  __syncthreads();
  for (int off = 128; off > 0; off >>= 1) {
    if (tid < off) { s1[tid] += s1[tid+off]; s2[tid] += s2[tid+off]; }
    __syncthreads();
  }
  float mu  = s1[0] * (1.f/DMODEL);
  float var = s2[0] * (1.f/DMODEL) - mu*mu;
  float inv = rsqrtf(var + LN_EPS);
  out[(size_t)tok*DMODEL + tid]       = gamma[tid]    *(v0-mu)*inv + beta[tid];
  out[(size_t)tok*DMODEL + tid + 256] = gamma[tid+256]*(v1-mu)*inv + beta[tid+256];
}

extern "C" void kernel_launch(void* const* d_in, const int* in_sizes, int n_in,
                              void* d_out, int out_size, void* d_ws, size_t ws_size,
                              hipStream_t stream) {
  const float* x   = (const float*)d_in[0];
  const float* lar = (const float*)d_in[1];
  const float* aim = (const float*)d_in[2];
  const float* Bp  = (const float*)d_in[3];
  const float* Cp  = (const float*)d_in[4];
  const float* Dp  = (const float*)d_in[5];
  const float* Wo  = (const float*)d_in[6];
  const float* bo  = (const float*)d_in[7];
  const float* gm  = (const float*)d_in[8];
  const float* bt  = (const float*)d_in[9];
  float* out = (float*)d_out;
  char*  ws  = (char*)d_ws;

  // workspace layout (bytes)
  const size_t OFF_KS = 0;                                  // 512*2048*2  = 2 MiB
  const size_t OFF_W  = OFF_KS + (size_t)DMODEL*SEQ*2;      // 512*512*2   = 0.5 MiB
  const size_t OFF_Y1 = OFF_W  + (size_t)DMODEL*DMODEL*2;   // 8192*512*2  = 8 MiB
  const size_t OFF_Z  = OFF_Y1 + (size_t)BATCH*SEQ*DMODEL*2;// 8192*512*4  = 16 MiB
  __bf16* Ksbf = (__bf16*)(ws + OFF_KS);
  __bf16* Wbf  = (__bf16*)(ws + OFF_W);
  __bf16* y1bf = (__bf16*)(ws + OFF_Y1);
  float*  z    = (float*)(ws + OFF_Z);

  wcvt_kernel<<<(DMODEL*DMODEL + 255)/256, 256, 0, stream>>>(Wo, Wbf, DMODEL*DMODEL);
  ssmk_kernel<<<DMODEL, 256, 0, stream>>>(lar, aim, Bp, Cp, Ksbf);
  conv_kernel<<<dim3(DMODEL, BATCH), 256, 0, stream>>>(x, Ksbf, Dp, y1bf);
  gemm_kernel<<<(BATCH*SEQ/32)*(DMODEL/64)/8, 256, 0, stream>>>(y1bf, Wbf, bo, x, z);
  ln_kernel<<<BATCH*SEQ, 256, 0, stream>>>(z, gm, bt, out);
}
